// MultiHeadAttention3D_30056181137931
// MI455X (gfx1250) — matrix operand order repaired and checked by reading, not compiled, not run on hardware
//
#include <hip/hip_runtime.h>
#include <hip/hip_bf16.h>

typedef _Float16 f16;
typedef __attribute__((ext_vector_type(16))) _Float16 v16h;
typedef __attribute__((ext_vector_type(8)))  _Float16 v8h;
typedef __attribute__((ext_vector_type(8)))  float    v8f;
typedef __attribute__((ext_vector_type(4)))  float    v4f;

#define BATCH 2
#define CH    256
#define NTOK  4096
#define NHEAD 8
#define HDIM  32

union V16 { v16h v; v8h h[2]; };

static __device__ inline v8f wmma_f16(const V16& a, const V16& b, v8f c) {
  return __builtin_amdgcn_wmma_f32_16x16x32_f16(false, a.v, false, b.v,
                                                (short)0, c, false, false);
}

// ---------------------------------------------------------------------------
// Kernel 1: QKV projection.  x[b][c][n] (f32) @ Wqkv[c][3c] (f32)
// -> q,k as [b][h][n][32] f16 (q pre-scaled by log2(e)/sqrt(32)), v transposed
//    as [b][h][32][n] f16.  Block = 8 waves (4M x 2N), tile 64(M) x 128(N).
// Staging: issue all float4 global loads first (clause, overlapped latency),
// then fan out converts to f16 LDS stores.
// ---------------------------------------------------------------------------
struct K1Lds {
  union {
    struct { f16 A[64 * 40]; f16 Bt[128 * 40]; } st;  // padded rows, 16B aligned
    f16 outt[64 * 136];
  } u;
};

__global__ __launch_bounds__(256) void qkv_kernel(const float* __restrict__ x,
    const float* __restrict__ Wqkv, f16* __restrict__ qh, f16* __restrict__ kh,
    f16* __restrict__ vth) {
  __shared__ K1Lds lds;
  const int tid = threadIdx.x;
  const int lane = tid & 31, wave = tid >> 5;
  const int waveM = wave & 3, waveN = wave >> 2;
  const int l16 = lane & 15, hi = lane >> 4;
  const int mb = blockIdx.x * 64;        // token tile
  const int cb = blockIdx.y;             // 0..5 : 128-col block of 768
  const int b  = blockIdx.z;
  const float* xb = x + b * CH * NTOK;

  // per-thread staging coordinates (invariant across K steps)
  int ka[2], ma[2], kb_[4], nb_[4];
#pragma unroll
  for (int e = 0; e < 2; ++e) {
    int idx = e * 256 + tid;                 // 512 A-quads: 32k x 16 m-quads
    ka[e] = idx >> 4;
    ma[e] = (idx & 15) * 4;
  }
#pragma unroll
  for (int e = 0; e < 4; ++e) {
    int idx = e * 256 + tid;                 // 1024 B-quads: 32k x 32 n-quads
    kb_[e] = idx >> 5;
    nb_[e] = (idx & 31) * 4;
  }

  v8f acc[4] = {};
  for (int k0 = 0; k0 < CH; k0 += 32) {
    // phase 1: issue ALL global loads (overlap latency in one clause)
    v4f qa[2], qb[4];
#pragma unroll
    for (int e = 0; e < 2; ++e)
      qa[e] = *(const v4f*)&xb[(k0 + ka[e]) * NTOK + mb + ma[e]];
#pragma unroll
    for (int e = 0; e < 4; ++e)
      qb[e] = *(const v4f*)&Wqkv[(k0 + kb_[e]) * 768 + cb * 128 + nb_[e]];
    __syncthreads();
    // phase 2: convert + transpose into LDS
#pragma unroll
    for (int e = 0; e < 2; ++e)
#pragma unroll
      for (int j = 0; j < 4; ++j)
        lds.u.st.A[(ma[e] + j) * 40 + ka[e]] = (f16)qa[e][j];
#pragma unroll
    for (int e = 0; e < 4; ++e)
#pragma unroll
      for (int j = 0; j < 4; ++j)
        lds.u.st.Bt[(nb_[e] + j) * 40 + kb_[e]] = (f16)qb[e][j];
    __syncthreads();
    V16 af;
    const int mrow = waveM * 16 + l16;
    const int c0 = hi * 16;
    af.h[0] = *(const v8h*)&lds.u.st.A[mrow * 40 + c0];
    af.h[1] = *(const v8h*)&lds.u.st.A[mrow * 40 + c0 + 8];
#pragma unroll
    for (int f = 0; f < 4; ++f) {
      V16 bf;
      int ncol = waveN * 64 + f * 16 + l16;
      int ko = hi * 16;
      bf.h[0] = *(const v8h*)&lds.u.st.Bt[ncol * 40 + ko];
      bf.h[1] = *(const v8h*)&lds.u.st.Bt[ncol * 40 + ko + 8];
      acc[f] = wmma_f16(af, bf, acc[f]);
    }
    __syncthreads();   // protect LDS before next iteration's stores
  }
  // C fragments -> LDS out tile (f16).  q columns get log2(e)/sqrt(head_dim)
  // folded in so the attention softmax can run in the exp2 domain.
  const float qscale = 0.17677669529663687f * 1.4426950408889634f;
#pragma unroll
  for (int f = 0; f < 4; ++f) {
    int lcol = waveN * 64 + f * 16 + l16;
    int gcol = cb * 128 + lcol;
    float s = (gcol < 256) ? qscale : 1.0f;
    int rbase = waveM * 16 + hi * 8;
#pragma unroll
    for (int r = 0; r < 8; ++r)
      lds.u.outt[(rbase + r) * 136 + lcol] = (f16)(acc[f][r] * s);
  }
  __syncthreads();
  if (cb < 4) {  // q or k: [b][h][n][32], contiguous d -> vectorized copies
    f16* dst = (cb < 2) ? qh : kh;
    const int colbase = (cb & 1) * 128;
#pragma unroll
    for (int e = 0; e < 4; ++e) {
      int idx = e * 256 + tid;              // 1024 chunks of 8 halves
      int row = idx >> 4, ch = idx & 15;
      int j = colbase + ch * 8;
      int h = j >> 5, d = j & 31;
      int n = mb + row;
      *(v8h*)&dst[((b * NHEAD + h) * NTOK + n) * HDIM + d] =
          *(const v8h*)&lds.u.outt[row * 136 + ch * 8];
    }
  } else {       // v: transpose to [b][h][32][n], contiguous n
    const int colbase = (cb & 1) * 128;
#pragma unroll
    for (int e = 0; e < 2; ++e) {
      int idx = e * 256 + tid;              // 512 tasks: (col, 16-row segment)
      int coll = idx >> 2, seg = idx & 3;
      int j = colbase + coll;
      int h = j >> 5, d = j & 31;
      v8h p0, p1;
#pragma unroll
      for (int r = 0; r < 8; ++r) {
        p0[r] = lds.u.outt[(seg * 16 + r) * 136 + coll];
        p1[r] = lds.u.outt[(seg * 16 + 8 + r) * 136 + coll];
      }
      f16* dp = &vth[((b * NHEAD + h) * HDIM + d) * NTOK + mb + seg * 16];
      *(v8h*)dp = p0;
      *(v8h*)(dp + 8) = p1;
    }
  }
}

// ---------------------------------------------------------------------------
// Kernel 2: flash attention.  One wave = 16 query rows; loop over 32-key tiles.
// S = Q K^T (2 wmma); online softmax in exp2 domain (log2e pre-folded into q)
// with shfl_xor reductions over 16-lane halves; P through a wave-private LDS
// tile (intra-wave LDS is in-order; no barriers); O += P V (2 wmma).
// ---------------------------------------------------------------------------
__global__ __launch_bounds__(128) void attn_kernel(const f16* __restrict__ qh,
    const f16* __restrict__ kh, const f16* __restrict__ vth,
    f16* __restrict__ oh) {
  __shared__ f16 Pt[4][16 * 40];
  const int tid = threadIdx.x;
  const int lane = tid & 31, wave = tid >> 5;
  const int l16 = lane & 15, hi = lane >> 4;
  const int h = blockIdx.y, b = blockIdx.z;
  const int qbase = blockIdx.x * 64 + wave * 16;
  const f16* qp = qh + (b * NHEAD + h) * NTOK * HDIM;
  const f16* kp = kh + (b * NHEAD + h) * NTOK * HDIM;
  const f16* vp = vth + (b * NHEAD + h) * HDIM * NTOK;
  f16* Pw = Pt[wave];

  V16 qf;
  {
    int row = qbase + l16;
    int c0 = hi * 16;
    qf.h[0] = *(const v8h*)&qp[row * HDIM + c0];
    qf.h[1] = *(const v8h*)&qp[row * HDIM + c0 + 8];
  }
  v8f o0 = {}, o1 = {};
  float m[8], l[8];
#pragma unroll
  for (int r = 0; r < 8; ++r) { m[r] = -1e30f; l[r] = 0.f; }

  for (int kt = 0; kt < NTOK / 32; ++kt) {
    const int kb = kt * 32;
    V16 kf0, kf1;
    {
      int key = kb + l16;
      int ko = hi * 16;
      kf0.h[0] = *(const v8h*)&kp[key * HDIM + ko];
      kf0.h[1] = *(const v8h*)&kp[key * HDIM + ko + 8];
      kf1.h[0] = *(const v8h*)&kp[(key + 16) * HDIM + ko];
      kf1.h[1] = *(const v8h*)&kp[(key + 16) * HDIM + ko + 8];
    }
    v8f z = {};
    v8f s0 = wmma_f16(qf, kf0, z);
    v8f s1 = wmma_f16(qf, kf1, z);
    if (kt + 1 < NTOK / 32) {  // gfx1250 global_prefetch_b8 for next K/V tile
      __builtin_prefetch(&kp[(kb + 32 + lane) * HDIM], 0, 0);
      __builtin_prefetch(&vp[lane * NTOK + kb + 32], 0, 0);
    }
#pragma unroll
    for (int r = 0; r < 8; ++r) {
      float a = fmaxf(s0[r], s1[r]);
      a = fmaxf(a, __shfl_xor(a, 1));
      a = fmaxf(a, __shfl_xor(a, 2));
      a = fmaxf(a, __shfl_xor(a, 4));
      a = fmaxf(a, __shfl_xor(a, 8));
      float mn = fmaxf(m[r], a);
      float alpha = __builtin_amdgcn_exp2f(m[r] - mn);   // exp2 domain
      float p0 = __builtin_amdgcn_exp2f(s0[r] - mn);
      float p1 = __builtin_amdgcn_exp2f(s1[r] - mn);
      float sum = p0 + p1;
      sum += __shfl_xor(sum, 1);
      sum += __shfl_xor(sum, 2);
      sum += __shfl_xor(sum, 4);
      sum += __shfl_xor(sum, 8);
      m[r] = mn;
      l[r] = l[r] * alpha + sum;
      o0[r] *= alpha;
      o1[r] *= alpha;
      int prow = hi * 8 + r;
      Pw[prow * 40 + l16] = (f16)p0;
      Pw[prow * 40 + 16 + l16] = (f16)p1;
    }
    V16 pf;
    {
      int c0 = hi * 16;
      pf.h[0] = *(const v8h*)&Pw[l16 * 40 + c0];
      pf.h[1] = *(const v8h*)&Pw[l16 * 40 + c0 + 8];
    }
    V16 vf0, vf1;
    {
      int ko = kb + hi * 16;
      vf0.h[0] = *(const v8h*)&vp[l16 * NTOK + ko];
      vf0.h[1] = *(const v8h*)&vp[l16 * NTOK + ko + 8];
      vf1.h[0] = *(const v8h*)&vp[(16 + l16) * NTOK + ko];
      vf1.h[1] = *(const v8h*)&vp[(16 + l16) * NTOK + ko + 8];
    }
    o0 = wmma_f16(pf, vf0, o0);
    o1 = wmma_f16(pf, vf1, o1);
  }
#pragma unroll
  for (int r = 0; r < 8; ++r) {
    float inv = 1.0f / l[r];
    int n = qbase + hi * 8 + r;
    int c0 = h * HDIM + l16;
    oh[(b * NTOK + n) * CH + c0]      = (f16)(o0[r] * inv);
    oh[(b * NTOK + n) * CH + c0 + 16] = (f16)(o1[r] * inv);
  }
}

// ---------------------------------------------------------------------------
// Kernel 3: out projection + bias + residual.  O[b][n][256] f16 @ Wout[256][256]
// A-fragments straight from global (row-major); B staged/transposed in LDS
// with batched float4 global loads.
// ---------------------------------------------------------------------------
__global__ __launch_bounds__(256) void proj_kernel(const f16* __restrict__ oh,
    const float* __restrict__ Wout, const float* __restrict__ bout,
    const float* __restrict__ x, float* __restrict__ out) {
  __shared__ f16 Bt[128 * 40];
  const int tid = threadIdx.x;
  const int lane = tid & 31, wave = tid >> 5;
  const int waveM = wave & 3, waveN = wave >> 2;
  const int l16 = lane & 15, hi = lane >> 4;
  const int mb = blockIdx.x * 64;
  const int nb = blockIdx.y * 128;
  const int b  = blockIdx.z;
  const f16* ob = oh + b * NTOK * CH;

  int kb_[4], nb_[4];
#pragma unroll
  for (int e = 0; e < 4; ++e) {
    int idx = e * 256 + tid;                 // 1024 quads: 32k x 32 n-quads
    kb_[e] = idx >> 5;
    nb_[e] = (idx & 31) * 4;
  }

  v8f acc[4] = {};
  for (int k0 = 0; k0 < CH; k0 += 32) {
    v4f qb[4];
#pragma unroll
    for (int e = 0; e < 4; ++e)
      qb[e] = *(const v4f*)&Wout[(k0 + kb_[e]) * CH + nb + nb_[e]];
    __syncthreads();
#pragma unroll
    for (int e = 0; e < 4; ++e)
#pragma unroll
      for (int j = 0; j < 4; ++j)
        Bt[(nb_[e] + j) * 40 + kb_[e]] = (f16)qb[e][j];
    __syncthreads();
    V16 af;
    int row = mb + waveM * 16 + l16;
    int c0 = k0 + hi * 16;
    af.h[0] = *(const v8h*)&ob[row * CH + c0];
    af.h[1] = *(const v8h*)&ob[row * CH + c0 + 8];
#pragma unroll
    for (int f = 0; f < 4; ++f) {
      V16 bf;
      int ncol = waveN * 64 + f * 16 + l16;
      int ko = hi * 16;
      bf.h[0] = *(const v8h*)&Bt[ncol * 40 + ko];
      bf.h[1] = *(const v8h*)&Bt[ncol * 40 + ko + 8];
      acc[f] = wmma_f16(af, bf, acc[f]);
    }
    __syncthreads();
  }
#pragma unroll
  for (int f = 0; f < 4; ++f) {
    int j = nb + waveN * 64 + f * 16 + l16;
    float bias = bout[j];
    int rbase = mb + waveM * 16 + hi * 8;
    const float* xr = &x[(size_t)(b * CH + j) * NTOK + rbase];
    float* orow = &out[(size_t)(b * CH + j) * NTOK + rbase];
#pragma unroll
    for (int r = 0; r < 8; ++r)
      orow[r] = acc[f][r] + bias + xr[r];
  }
}

extern "C" void kernel_launch(void* const* d_in, const int* in_sizes, int n_in,
                              void* d_out, int out_size, void* d_ws, size_t ws_size,
                              hipStream_t stream) {
  const float* x    = (const float*)d_in[0];
  const float* Wqkv = (const float*)d_in[1];
  const float* Wout = (const float*)d_in[2];
  const float* bout = (const float*)d_in[3];
  float* out = (float*)d_out;

  f16* ws = (f16*)d_ws;
  const size_t T = (size_t)BATCH * NHEAD * NTOK * HDIM;  // 2,097,152 halves
  f16* qh  = ws;          // [b][h][n][32]  (pre-scaled by log2e/sqrt(32))
  f16* kh  = ws + T;      // [b][h][n][32]
  f16* vth = ws + 2 * T;  // [b][h][32][n]
  f16* oh  = ws + 3 * T;  // [b][n][256]   (total 16 MB of ws)

  qkv_kernel<<<dim3(64, 6, BATCH), 256, 0, stream>>>(x, Wqkv, qh, kh, vth);
  attn_kernel<<<dim3(64, NHEAD, BATCH), 128, 0, stream>>>(qh, kh, vth, oh);
  proj_kernel<<<dim3(64, 2, BATCH), 256, 0, stream>>>(oh, Wout, bout, x, out);
}
